// ArcFace_84078279786622
// MI455X (gfx1250) — matrix operand order repaired and checked by reading, not compiled, not run on hardware
//
#include <hip/hip_runtime.h>
#include <math.h>

#define BATCH 512
#define FEAT  512
#define NCLS  100000
#define NT    128                       // columns per workgroup
#define CB    ((NCLS + NT - 1) / NT)    // 782 column blocks
#define LCS   520                       // LDS column stride in bf16 units (512 + 8 pad = 16B pad)

typedef __attribute__((ext_vector_type(16))) __bf16        v16bf;
typedef __attribute__((ext_vector_type(8)))  float         v8f;
typedef __attribute__((ext_vector_type(4)))  unsigned int  u32x4;

static __device__ __forceinline__ unsigned short f2bf(float f) {
  union { float f; unsigned int u; } c; c.f = f;
  unsigned int u = c.u;
  u += 0x7fffu + ((u >> 16) & 1u);      // round-to-nearest-even
  return (unsigned short)(u >> 16);
}
static __device__ __forceinline__ float bf2f(unsigned short h) {
  union { unsigned int u; float f; } c; c.u = ((unsigned int)h) << 16;
  return c.f;
}
// Fragment = 16 bf16 = two contiguous 16-byte runs (per ISA 16-bit A/B VGPR layouts)
static __device__ __forceinline__ v16bf load_frag(const unsigned short* p0,
                                                  const unsigned short* p1) {
  v16bf r;
  ((u32x4*)&r)[0] = *(const u32x4*)p0;
  ((u32x4*)&r)[1] = *(const u32x4*)p1;
  return r;
}

// ---------------------------------------------------------------- normalize
// x_n = x / ||x||_row ; split into bf16 hi + lo residual.
__global__ void __launch_bounds__(FEAT) arc_normalize(
    const float* __restrict__ x, unsigned short* __restrict__ xhi,
    unsigned short* __restrict__ xlo) {
  __shared__ float red[FEAT];
  const int r = blockIdx.x, t = threadIdx.x;
  const float v = x[r * FEAT + t];
  red[t] = v * v;
  __syncthreads();
  for (int s = FEAT / 2; s > 0; s >>= 1) {
    if (t < s) red[t] += red[t + s];
    __syncthreads();
  }
  const float rn = 1.0f / sqrtf(red[0]);
  const float xn = v * rn;
  const unsigned short h = f2bf(xn);
  xhi[r * FEAT + t] = h;
  xlo[r * FEAT + t] = f2bf(xn - bf2f(h));
}

// ---------------------------------------------------------------- GEMM + online softmax partials
// Each block: 128-column stripe of w staged in LDS (bf16 hi/lo, 260 KB),
// 8 waves x 16 columns. Two row tiles per iteration with two independent
// accumulator chains (WMMA ILP) sharing the B fragments; 3 bf16 WMMAs per
// K-chunk per tile (hi*hi + hi*lo + lo*hi) into f32 accumulators.
__global__ void __launch_bounds__(256) arc_gemm(
    const float* __restrict__ w, const unsigned short* __restrict__ xhi,
    const unsigned short* __restrict__ xlo, const int* __restrict__ y,
    float* __restrict__ cosy, float* __restrict__ pe, float* __restrict__ pl) {
  extern __shared__ char smem[];
  unsigned short* bh = (unsigned short*)smem;          // NT * LCS bf16
  unsigned short* bl = bh + NT * LCS;                  // NT * LCS bf16
  float* red = (float*)(bl + NT * LCS);                // 8 waves * 32 rows * 2

  const int tid  = threadIdx.x;
  const int lane = tid & 31;
  const int wv   = tid >> 5;
  const int c0   = blockIdx.x * NT;

  // Stage w tile: fp32 -> bf16 hi + lo (coalesced: consecutive tid -> consecutive column)
  for (int i = tid; i < NT * FEAT; i += 256) {
    const int col = i & (NT - 1);
    const int k   = i >> 7;
    const int c   = c0 + col;
    const float v = (c < NCLS) ? w[(size_t)k * NCLS + c] : 0.0f;
    const unsigned short h = f2bf(v);
    bh[col * LCS + k] = h;
    bl[col * LCS + k] = f2bf(v - bf2f(h));
  }
  __syncthreads();

  const int  nloc = lane & 15;
  const int  colg = c0 + wv * 16 + nloc;
  const bool valid = colg < NCLS;
  const unsigned short* bch = bh + (wv * 16 + nloc) * LCS;
  const unsigned short* bcl = bl + (wv * 16 + nloc) * LCS;
  const int dA = (lane >> 4) * 8;    // A: upper half-wave holds K+8  (16-bit A layout)
  const int dB = (lane >> 4) * 8;    // B: upper half-wave holds K+8  (same order as A)

  for (int m = 0; m < BATCH / 16; m += 2) {
    const int rA0 = m * 16 + (lane & 15);
    const unsigned short* ah0 = xhi + rA0 * FEAT;
    const unsigned short* al0 = xlo + rA0 * FEAT;
    const unsigned short* ah1 = ah0 + 16 * FEAT;
    const unsigned short* al1 = al0 + 16 * FEAT;
    v8f acc0 = {};
    v8f acc1 = {};
    for (int kc = 0; kc < FEAT / 32; ++kc) {
      const int k0 = kc * 32;
      const v16bf fbh  = load_frag(bch + k0 + dB, bch + k0 + 16 + dB);
      const v16bf fbl  = load_frag(bcl + k0 + dB, bcl + k0 + 16 + dB);
      const v16bf fah0 = load_frag(ah0 + k0 + dA, ah0 + k0 + 16 + dA);
      const v16bf fal0 = load_frag(al0 + k0 + dA, al0 + k0 + 16 + dA);
      const v16bf fah1 = load_frag(ah1 + k0 + dA, ah1 + k0 + 16 + dA);
      const v16bf fal1 = load_frag(al1 + k0 + dA, al1 + k0 + 16 + dA);
      // Two independent accumulator chains interleaved -> no back-to-back
      // RAW on the same WMMA destination.
      acc0 = __builtin_amdgcn_wmma_f32_16x16x32_bf16(false, fah0, false, fbh,
                                                     (short)0, acc0, false, false);
      acc1 = __builtin_amdgcn_wmma_f32_16x16x32_bf16(false, fah1, false, fbh,
                                                     (short)0, acc1, false, false);
      acc0 = __builtin_amdgcn_wmma_f32_16x16x32_bf16(false, fah0, false, fbl,
                                                     (short)0, acc0, false, false);
      acc1 = __builtin_amdgcn_wmma_f32_16x16x32_bf16(false, fah1, false, fbl,
                                                     (short)0, acc1, false, false);
      acc0 = __builtin_amdgcn_wmma_f32_16x16x32_bf16(false, fal0, false, fbh,
                                                     (short)0, acc0, false, false);
      acc1 = __builtin_amdgcn_wmma_f32_16x16x32_bf16(false, fal1, false, fbh,
                                                     (short)0, acc1, false, false);
    }

    // Epilogue (both tiles): fixed-max (=S) online softmax partials + gather cos[y].
    union { v8f v; float f[8]; } ua0, ua1;
    ua0.v = acc0; ua1.v = acc1;
    for (int t = 0; t < 2; ++t) {
      const float* af = t ? ua1.f : ua0.f;
      for (int vv = 0; vv < 8; ++vv) {
        const float cosv = af[vv];
        float lg = valid ? (64.0f * cosv) : 0.0f;
        float ee = valid ? __expf(64.0f * cosv - 64.0f) : 0.0f;
        const int rloc = (lane < 16) ? vv : (8 + vv);  // C/D layout: VGPR v -> M=v / M=8+v
        const int R = m * 16 + t * 16 + rloc;
        if (valid && y[R] == colg) cosy[R] = cosv;
        for (int off = 8; off; off >>= 1) {            // reduce within each 16-lane half (N dim)
          ee += __shfl_xor(ee, off, 32);
          lg += __shfl_xor(lg, off, 32);
        }
        if ((lane & 15) == 0) {
          red[(wv * 32 + t * 16 + rloc) * 2 + 0] = ee;
          red[(wv * 32 + t * 16 + rloc) * 2 + 1] = lg;
        }
      }
    }
    __syncthreads();
    if (tid < 32) {                                    // combine 8 waves, fixed order
      float se = 0.0f, sl = 0.0f;
      for (int wq = 0; wq < 8; ++wq) {
        se += red[(wq * 32 + tid) * 2 + 0];
        sl += red[(wq * 32 + tid) * 2 + 1];
      }
      const int R = m * 16 + tid;
      pe[blockIdx.x * BATCH + R] = se;
      pl[blockIdx.x * BATCH + R] = sl;
    }
    __syncthreads();
  }
}

// ---------------------------------------------------------------- finalize
__global__ void __launch_bounds__(BATCH) arc_finalize(
    const float* __restrict__ pe, const float* __restrict__ pl,
    const float* __restrict__ cosy, float* __restrict__ out) {
  __shared__ float red[BATCH];
  const int r = threadIdx.x;
  float se = 0.0f, sl = 0.0f;
  for (int cb = 0; cb < CB; ++cb) {      // coalesced, deterministic order
    se += pe[cb * BATCH + r];
    sl += pl[cb * BATCH + r];
  }
  const float cy = cosy[r];
  float cc = fminf(fmaxf(cy, -1.0f + 1e-5f), 1.0f - 1e-5f);
  float th = acosf(cc) + 0.5f;
  th = fminf(fmaxf(th, 1e-5f), 3.14159f);
  const float cm = cosf(th);
  se += __expf(64.0f * cm - 64.0f) - __expf(64.0f * cy - 64.0f);  // margin fixup at y
  sl += 64.0f * cm - 64.0f * cy;
  const float lse    = 64.0f + logf(se);
  const float nll    = lse - 64.0f * cm;
  const float smooth = lse - sl * (1.0f / (float)NCLS);
  red[r] = 0.9f * nll + 0.1f * smooth;
  __syncthreads();
  for (int s = BATCH / 2; s > 0; s >>= 1) {
    if (r < s) red[r] += red[r + s];
    __syncthreads();
  }
  if (r == 0) out[0] = red[0] * (1.0f / (float)BATCH);
}

// ---------------------------------------------------------------- launch
extern "C" void kernel_launch(void* const* d_in, const int* in_sizes, int n_in,
                              void* d_out, int out_size, void* d_ws, size_t ws_size,
                              hipStream_t stream) {
  const float* x = (const float*)d_in[0];
  const int*   y = (const int*)d_in[1];
  // d_in[2] = class_freq: unused (balanced=False)
  const float* w = (const float*)d_in[3];
  float* out = (float*)d_out;

  char* ws = (char*)d_ws;
  unsigned short* xhi = (unsigned short*)(ws);                       // 512 KB
  unsigned short* xlo = (unsigned short*)(ws + 524288);              // 512 KB
  float* cosy = (float*)(ws + 1048576);                              // 2 KB
  float* pe   = (float*)(ws + 1050624);                              // CB*512*4 = 1.6 MB
  float* pl   = (float*)(ws + 1050624 + (size_t)CB * BATCH * 4);     // 1.6 MB

  arc_normalize<<<BATCH, FEAT, 0, stream>>>(x, xhi, xlo);

  const size_t lds_bytes = 2u * NT * LCS * sizeof(unsigned short)    // 266240
                         + 8u * 32u * 2u * sizeof(float);            // + 2048
  arc_gemm<<<CB, 256, lds_bytes, stream>>>(w, xhi, xlo, y, cosy, pe, pl);

  arc_finalize<<<1, BATCH, 0, stream>>>(pe, pl, cosy, out);
}
